// QuadraticAttention_82094004896066
// MI455X (gfx1250) — matrix operand order repaired and checked by reading, not compiled, not run on hardware
//
#include <hip/hip_runtime.h>

typedef unsigned short u16t;
typedef unsigned int   u32t;
typedef __attribute__((ext_vector_type(16))) __bf16 v16bf;
typedef __attribute__((ext_vector_type(8)))  float  v8f;
typedef __attribute__((ext_vector_type(8)))  u32t   v8u;
typedef __attribute__((ext_vector_type(4)))  int    i32x4;

// ---------- CDNA5 async global->LDS staging (ASYNCcnt path) ----------
#if defined(__has_builtin)
#if __has_builtin(__builtin_amdgcn_global_load_async_to_lds_b128)
#define USE_ASYNC_LDS 1
#endif
#endif

#if defined(USE_ASYNC_LDS)
#if __has_builtin(__builtin_amdgcn_s_wait_asynccnt)
#define WAIT_ASYNC() __builtin_amdgcn_s_wait_asynccnt(0)
#else
#define WAIT_ASYNC() asm volatile("s_wait_asynccnt 0x0" ::: "memory")
#endif

__device__ inline void async_cp128(const u16t* g, u16t* l) {
    // 16 bytes per lane, global -> LDS, tracked by ASYNCcnt.
    // Builtin signature (from probe): (i32x4 AS1*, i32x4 AS3*, imm, imm), non-const.
    __attribute__((address_space(1))) i32x4* gp =
        (__attribute__((address_space(1))) i32x4*)(unsigned long long)g;
    __attribute__((address_space(3))) i32x4* lp =
        (__attribute__((address_space(3))) i32x4*)(u32t)(unsigned long long)l;
    __builtin_amdgcn_global_load_async_to_lds_b128(gp, lp, 0, 0);
}
#endif

// ---------- helpers ----------
__device__ inline u16t f2bf(float f) {
    u32t u = __builtin_bit_cast(u32t, f);
    u32t r = u + 0x7FFFu + ((u >> 16) & 1u);   // round-to-nearest-even
    return (u16t)(r >> 16);
}

__device__ inline v16bf make_frag(uint4 c0, uint4 c1) {
    v8u u;
    u[0] = c0.x; u[1] = c0.y; u[2] = c0.z; u[3] = c0.w;
    u[4] = c1.x; u[5] = c1.y; u[6] = c1.z; u[7] = c1.w;
    return __builtin_bit_cast(v16bf, u);
}

// A-fragment (16x32 bf16): lane m = L&15, kb = (L>>4)*8.
// halfs 0..7 = K kb..kb+7 ; halfs 8..15 = K kb+16..kb+23 (ISA 7.12.2)
__device__ inline v16bf load_afrag(const u16t* base, int row_stride, int lane, int koff) {
    int m  = lane & 15;
    int kb = (lane >> 4) << 3;
    const u16t* p = base + (size_t)m * row_stride + koff + kb;
    uint4 c0 = *(const uint4*)(p);
    uint4 c1 = *(const uint4*)(p + 16);
    return make_frag(c0, c1);
}

// B-fragment (32x16 bf16): lane n = L&15, kb = (L>>4)*8; halfs 0..7 = K kb..kb+7, halfs 8..15 = K kb+16..kb+23
__device__ inline v16bf load_bfrag(const u16t* base, int row_stride, int lane) {
    int n  = lane & 15;
    int kb = (lane >> 4) << 3;
    v16bf r;
#pragma unroll
    for (int s = 0; s < 8; ++s) {
        r[s]     = __builtin_bit_cast(__bf16, base[(size_t)(kb + s) * row_stride + n]);
        r[s + 8] = __builtin_bit_cast(__bf16, base[(size_t)(kb + 16 + s) * row_stride + n]);
    }
    return r;
}

__device__ inline v8f wmma_bf16(v16bf a, v16bf b, v8f c) {
    return __builtin_amdgcn_wmma_f32_16x16x32_bf16(false, a, false, b, (short)0, c, false, false);
}

// ---------- f32 -> bf16 convert ----------
__global__ void cvt_bf16_kernel(const float* __restrict__ in, u16t* __restrict__ out, int n) {
    int i = blockIdx.x * blockDim.x + threadIdx.x;
    if (i < n) out[i] = f2bf(in[i]);
}

// ---------- generic bf16 WMMA GEMM: C[MxN] = A[MxK] @ B[KxN] ----------
// WG tile 64x64, 4 waves, each wave 32x32 (2x2 WMMA tiles), K step 32.
__global__ __launch_bounds__(128)
void gemm_bf16_kernel(const u16t* __restrict__ A, const u16t* __restrict__ B,
                      float* __restrict__ Cf, u16t* __restrict__ Cb,
                      int M, int N, int K)
{
    __shared__ u16t lA[64 * 40];   // 64 rows x 32 cols, padded stride 40
    __shared__ u16t lB[32 * 72];   // 32 rows x 64 cols, padded stride 72

    const int lane = threadIdx.x & 31;
    const int wave = threadIdx.x >> 5;
    const int wm   = (wave & 1) * 32;
    const int wn   = (wave >> 1) * 32;
    const int wgM  = blockIdx.y * 64;
    const int wgN  = blockIdx.x * 64;

    v8f acc[2][2] = {};

    for (int k0 = 0; k0 < K; k0 += 32) {
#if defined(USE_ASYNC_LDS)
        {   // A tile: async global->LDS, 32 B per thread
            int r = threadIdx.x >> 1;
            int h = (threadIdx.x & 1) * 16;
            const u16t* g = A + (size_t)(wgM + r) * K + k0 + h;
            async_cp128(g,     &lA[r * 40 + h]);
            async_cp128(g + 8, &lA[r * 40 + h + 8]);
        }
        {   // B tile
            int r = threadIdx.x >> 2;
            int q = (threadIdx.x & 3) * 16;
            const u16t* g = B + (size_t)(k0 + r) * N + wgN + q;
            async_cp128(g,     &lB[r * 72 + q]);
            async_cp128(g + 8, &lB[r * 72 + q + 8]);
        }
        WAIT_ASYNC();
        __syncthreads();
#else
        {   // A tile via VGPR bounce (fallback)
            int r = threadIdx.x >> 1;
            int h = (threadIdx.x & 1) * 16;
            const u16t* g = A + (size_t)(wgM + r) * K + k0 + h;
            uint4 d0 = *(const uint4*)(g);
            uint4 d1 = *(const uint4*)(g + 8);
            *(uint4*)(&lA[r * 40 + h])     = d0;
            *(uint4*)(&lA[r * 40 + h + 8]) = d1;
        }
        {   // B tile
            int r = threadIdx.x >> 2;
            int q = (threadIdx.x & 3) * 16;
            const u16t* g = B + (size_t)(k0 + r) * N + wgN + q;
            uint4 d0 = *(const uint4*)(g);
            uint4 d1 = *(const uint4*)(g + 8);
            *(uint4*)(&lB[r * 72 + q])     = d0;
            *(uint4*)(&lB[r * 72 + q + 8]) = d1;
        }
        __syncthreads();
#endif

        v16bf a0 = load_afrag(&lA[(wm +  0) * 40], 40, lane, 0);
        v16bf a1 = load_afrag(&lA[(wm + 16) * 40], 40, lane, 0);
        v16bf b0 = load_bfrag(&lB[wn +  0], 72, lane);
        v16bf b1 = load_bfrag(&lB[wn + 16], 72, lane);

        acc[0][0] = wmma_bf16(a0, b0, acc[0][0]);
        acc[0][1] = wmma_bf16(a0, b1, acc[0][1]);
        acc[1][0] = wmma_bf16(a1, b0, acc[1][0]);
        acc[1][1] = wmma_bf16(a1, b1, acc[1][1]);
        __syncthreads();
    }

    // D layout: n = lane&15, m = (lane>>4)*8 + r
    int n  = lane & 15;
    int rb = (lane >> 4) * 8;
#pragma unroll
    for (int mt = 0; mt < 2; ++mt)
#pragma unroll
        for (int nt = 0; nt < 2; ++nt)
#pragma unroll
            for (int r = 0; r < 8; ++r) {
                size_t gm = wgM + wm + mt * 16 + rb + r;
                size_t gn = wgN + wn + nt * 16 + n;
                float val = acc[mt][nt][r];
                if (Cb) Cb[gm * N + gn] = f2bf(val);
                else    Cf[gm * N + gn] = val;
            }
}

// ---------- fused causal (strict) attention, no softmax ----------
// One wave per (b, h, 16-row t tile). O[16x64] accumulated in f32.
__global__ __launch_bounds__(32)
void attn_kernel(const u16t* __restrict__ q, const u16t* __restrict__ k,
                 const u16t* __restrict__ v, u16t* __restrict__ o,
                 int T, int HD)
{
    __shared__ u16t lS[16 * 40];   // S tile: 16 t-rows x 32 s-cols, padded

    const int lane = threadIdx.x;
    const int tt   = blockIdx.x;
    const int h    = blockIdx.y;
    const int b    = blockIdx.z;
    const int t0   = tt * 16;
    const size_t rowbase = (size_t)b * T * HD + (size_t)h * 64;

    // q^T as two B-fragments (K = head dim): lane n = t, halfs = kk in table order.
    v16bf qb[2];
#pragma unroll
    for (int c = 0; c < 2; ++c) {
        qb[c] = load_afrag(q + rowbase + (size_t)t0 * HD, HD, lane, c * 32);
    }

    v8f o_acc[4] = {};

    // only s <= t0+14 can survive the strict mask
    const int nchunks = (t0 + 14) / 32 + 1;

    for (int ch = 0; ch < nchunks; ++ch) {
        const int s0 = ch * 32;

        if (s0 + 32 + lane < T)   // speculative prefetch of next k chunk
            __builtin_prefetch(k + rowbase + (size_t)(s0 + 32 + lane) * HD, 0, 1);

        // S^T = k @ q^T per 16-row s sub-tile; mask; store transposed bf16 to LDS
#pragma unroll
        for (int ss = 0; ss < 2; ++ss) {
            const u16t* krows = k + rowbase + (size_t)(s0 + ss * 16) * HD;
            v16bf ka0 = load_afrag(krows, HD, lane, 0);
            v16bf ka1 = load_afrag(krows, HD, lane, 32);
            v8f st = {};
            st = wmma_bf16(ka0, qb[0], st);
            st = wmma_bf16(ka1, qb[1], st);

            int tcol = lane & 15;          // n = t
            int srb  = (lane >> 4) * 8;    // m = s
            int tg   = t0 + tcol;
#pragma unroll
            for (int r = 0; r < 8; ++r) {
                int sl = srb + r;
                int sg = s0 + ss * 16 + sl;
                float val = (sg < tg) ? st[r] : 0.0f;   // strict lower-tri mask
                lS[tcol * 40 + ss * 16 + sl] = f2bf(val);
            }
        }
        __syncthreads();

        v16bf sa = load_afrag(lS, 40, lane, 0);   // S as 16x32 A-fragment
        __syncthreads();                          // lS reusable next chunk

        // v rows as B-fragments: lane n = d, halfs = s in table order
        const u16t* vrow = v + rowbase + (size_t)s0 * HD;
#pragma unroll
        for (int nd = 0; nd < 4; ++nd) {
            o_acc[nd] = wmma_bf16(sa, load_bfrag(vrow + nd * 16, HD, lane), o_acc[nd]);
        }
    }

    // store O (bf16) for the final Wo GEMM
    int dcol = lane & 15;
    int rb   = (lane >> 4) * 8;
#pragma unroll
    for (int nd = 0; nd < 4; ++nd)
#pragma unroll
        for (int r = 0; r < 8; ++r) {
            int tg = t0 + rb + r;
            o[rowbase + (size_t)tg * HD + nd * 16 + dcol] = f2bf(o_acc[nd][r]);
        }
}

// ---------- launch ----------
extern "C" void kernel_launch(void* const* d_in, const int* in_sizes, int n_in,
                              void* d_out, int out_size, void* d_ws, size_t ws_size,
                              hipStream_t stream) {
    (void)in_sizes; (void)n_in; (void)out_size; (void)ws_size;

    const float* x  = (const float*)d_in[0];
    const float* Wq = (const float*)d_in[1];
    const float* Wk = (const float*)d_in[2];
    const float* Wv = (const float*)d_in[3];
    const float* Wo = (const float*)d_in[4];
    float* out = (float*)d_out;

    const int Bb = 2, T = 2048, D = 1024;
    const int BT = Bb * T;                 // 4096
    const size_t NX = (size_t)BT * D;      // 4,194,304
    const size_t NW = (size_t)D * D;       // 1,048,576

    u16t* ws  = (u16t*)d_ws;
    u16t* xb  = ws;
    u16t* wqb = xb  + NX;
    u16t* wkb = wqb + NW;
    u16t* wvb = wkb + NW;
    u16t* wob = wvb + NW;
    u16t* qb  = wob + NW;
    u16t* kb  = qb  + NX;
    u16t* vb  = kb  + NX;
    u16t* ab  = vb  + NX;

    // 1) convert inputs to bf16
    cvt_bf16_kernel<<<(int)((NX + 255) / 256), 256, 0, stream>>>(x,  xb,  (int)NX);
    cvt_bf16_kernel<<<(int)((NW + 255) / 256), 256, 0, stream>>>(Wq, wqb, (int)NW);
    cvt_bf16_kernel<<<(int)((NW + 255) / 256), 256, 0, stream>>>(Wk, wkb, (int)NW);
    cvt_bf16_kernel<<<(int)((NW + 255) / 256), 256, 0, stream>>>(Wv, wvb, (int)NW);
    cvt_bf16_kernel<<<(int)((NW + 255) / 256), 256, 0, stream>>>(Wo, wob, (int)NW);

    // 2) q,k,v projections (bf16 out)
    dim3 ggrid(D / 64, BT / 64, 1);
    gemm_bf16_kernel<<<ggrid, 128, 0, stream>>>(xb, wqb, nullptr, qb, BT, D, D);
    gemm_bf16_kernel<<<ggrid, 128, 0, stream>>>(xb, wkb, nullptr, kb, BT, D, D);
    gemm_bf16_kernel<<<ggrid, 128, 0, stream>>>(xb, wvb, nullptr, vb, BT, D, D);

    // 3) fused strict-causal attention (bf16 out)
    attn_kernel<<<dim3(T / 16, 16, Bb), 32, 0, stream>>>(qb, kb, vb, ab, T, D);

    // 4) output projection (f32 out)
    gemm_bf16_kernel<<<ggrid, 128, 0, stream>>>(ab, wob, out, nullptr, BT, D, D);
}
